// DecoderLSTM_10849087389855
// MI455X (gfx1250) — matrix operand order repaired and checked by reading, not compiled, not run on hardware
//
#include <hip/hip_runtime.h>
#include <stdint.h>

// ---------------------------------------------------------------------------
// Types for CDNA5 WMMA (wave32, 16x16x32 bf16 -> f32)
// ---------------------------------------------------------------------------
typedef __attribute__((ext_vector_type(16))) __bf16 v16bf;
typedef __attribute__((ext_vector_type(8)))  float  v8f;

__device__ __forceinline__ uint16_t f2bf(float x) {
    uint32_t u = __float_as_uint(x);
    uint32_t r = u + 0x7FFFu + ((u >> 16) & 1u);   // round-to-nearest-even
    return (uint16_t)(r >> 16);
}
__device__ __forceinline__ float sigf(float x) { return 1.0f / (1.0f + __expf(-x)); }

// ---------------------------------------------------------------------------
// fp32 -> bf16 conversion
// ---------------------------------------------------------------------------
__global__ void cvt_bf16_kernel(const float* __restrict__ src, uint16_t* __restrict__ dst, int n) {
    int i = blockIdx.x * blockDim.x + threadIdx.x;
    if (i < n) dst[i] = f2bf(src[i]);
}

// K-concat two row-major f32 matrices [N,K1],[N,K2] into bf16 [N,K1+K2]
__global__ void cvt_concat_kernel(const float* __restrict__ s1, int K1,
                                  const float* __restrict__ s2, int K2,
                                  uint16_t* __restrict__ dst, int total) {
    int i = blockIdx.x * blockDim.x + threadIdx.x;
    if (i >= total) return;
    int Kt = K1 + K2;
    int n = i / Kt, k = i % Kt;
    float v = (k < K1) ? s1[(size_t)n * K1 + k] : s2[(size_t)n * K2 + (k - K1)];
    dst[i] = f2bf(v);
}

// concat two f32 bias vectors
__global__ void concat_bias_kernel(const float* __restrict__ a, int n1,
                                   const float* __restrict__ b, int n2,
                                   float* __restrict__ dst) {
    int i = blockIdx.x * blockDim.x + threadIdx.x;
    if (i < n1) dst[i] = a[i];
    else if (i < n1 + n2) dst[i] = b[i - n1];
}

// ---------------------------------------------------------------------------
// mean over L (original [B,F,L] layout), output bf16 [B,F]
// ---------------------------------------------------------------------------
__global__ void mean_kernel(const float* __restrict__ features, uint16_t* __restrict__ mean_bf) {
    int b = blockIdx.x, f = threadIdx.x;           // 64 blocks x 512 threads
    const float* p = features + ((size_t)b * 512 + f) * 196;
    float s = 0.f;
    for (int l = 0; l < 196; ++l) s += p[l];
    mean_bf[b * 512 + f] = f2bf(s * (1.0f / 196.0f));
}

// ---------------------------------------------------------------------------
// Generic WMMA GEMM:
//   C[M,N](f32, row stride ldc) = A[M,K](bf16, row stride lda) * W[N,K](bf16)^T
//                                 + bias + bias2
//   act==1 -> sigmoid on all cols; act==2 -> sigmoid on cols >= 512
//   Cbf (row stride ldcb) -> also store bf16 copy
//   lengths -> masked store: row b inactive if (lengths[b]-1) <= step => write 0
// One wave per 16x16 output tile; K in chunks of 32 (global_load_b128 pairs).
// ---------------------------------------------------------------------------
__global__ void wmma_gemm_kernel(const uint16_t* __restrict__ A, int lda,
                                 const uint16_t* __restrict__ W,
                                 const float* __restrict__ bias,
                                 const float* __restrict__ bias2,
                                 float* __restrict__ C, uint16_t* __restrict__ Cbf,
                                 int M, int N, int K, int ldc, int ldcb, int act,
                                 const int* __restrict__ lengths, int step) {
    int wave = (blockIdx.x * blockDim.x + threadIdx.x) >> 5;
    int mt = M >> 4;
    int total = mt * (N >> 4);
    if (wave >= total) return;                      // wave-uniform exit (EXEC all-ones for WMMA)
    int mtile = wave % mt;
    int ntile = wave / mt;

    int lane = threadIdx.x & 31;
    int half = lane >> 4;                            // 0: lanes 0-15, 1: lanes 16-31
    int lm   = lane & 15;

    int arow = (mtile << 4) + lm;                    // A row (M index) held by this lane
    int brow = (ntile << 4) + lm;                    // W row == output column held by this lane

    const uint4* Ap4 = (const uint4*)(A + (size_t)arow * lda);  // strides 16B aligned
    const uint4* Wp4 = (const uint4*)(W + (size_t)brow * K);

    v8f acc = {0.f, 0.f, 0.f, 0.f, 0.f, 0.f, 0.f, 0.f};

    for (int k0 = 0; k0 < K; k0 += 32) {
        union { v16bf v; uint4 q[2]; } av, bv;
        int base = k0 >> 3;                          // uint4 index of this 32-K chunk
        // A fragment (16-bit A 16x32 layout): lanes0-15 K=0..7,16..23; lanes16-31 K=8..15,24..31
        av.q[0] = Ap4[base + half];
        av.q[1] = Ap4[base + 2 + half];
        // B fragment (same K order as A): lanes0-15 K=0..7,16..23; lanes16-31 K=8..15,24..31
        bv.q[0] = Wp4[base + half];
        bv.q[1] = Wp4[base + 2 + half];
        acc = __builtin_amdgcn_wmma_f32_16x16x32_bf16(false, av.v, false, bv.v,
                                                      (short)0, acc, false, false);
    }

    int col = (ntile << 4) + lm;
    float bval = (bias ? bias[col] : 0.f) + (bias2 ? bias2[col] : 0.f);
#pragma unroll
    for (int v = 0; v < 8; ++v) {
        int row = (mtile << 4) + v + (half << 3);    // C layout: VGPR v -> M=v (lo half) / v+8
        float val = acc[v] + bval;
        if (act == 1 || (act == 2 && col >= 512)) val = sigf(val);
        if (lengths && !((lengths[row] - 1) > step)) val = 0.f;
        C[(size_t)row * ldc + col] = val;
        if (Cbf) Cbf[(size_t)row * ldcb + col] = f2bf(val);
    }
}

// ---------------------------------------------------------------------------
// Attention: full = tanh(en_att + de) . W_full ; softmax over L ; z = feat^T alpha * fb
// df = [de | fb] fused buffer, row stride 1024 (fb already sigmoid'ed).
// Writes masked alphas and xh[:,0:512]=emb gather, xh[:,512:1024]=z (bf16, stride 1536).
// One block per batch row.
// ---------------------------------------------------------------------------
__global__ void attention_kernel(const float* __restrict__ en_att, const float* __restrict__ df,
                                 const float* __restrict__ W_full, const float* __restrict__ b_full,
                                 const float* __restrict__ features,
                                 const int* __restrict__ captions, const int* __restrict__ lengths,
                                 const uint16_t* __restrict__ Wemb_bf, uint16_t* __restrict__ xh,
                                 float* __restrict__ alphas_out, int t) {
    int b = blockIdx.x;
    int tid = threadIdx.x;                           // 256 threads
    __shared__ float s_de[512], s_wf[512], s_full[196], s_alpha[196], s_red[256];

    for (int i = tid; i < 512; i += 256) { s_de[i] = df[b * 1024 + i]; s_wf[i] = W_full[i]; }
    __syncthreads();

    for (int l = tid; l < 196; l += 256) {
        const float* row = en_att + ((size_t)(b * 196 + l)) * 512;
        float acc = 0.f;
        for (int a = 0; a < 512; ++a) acc += tanhf(row[a] + s_de[a]) * s_wf[a];
        s_full[l] = acc + b_full[0];
    }
    __syncthreads();

    // softmax over L=196
    float v = (tid < 196) ? s_full[tid] : -3.4e38f;
    s_red[tid] = v; __syncthreads();
    for (int off = 128; off > 0; off >>= 1) {
        if (tid < off) s_red[tid] = fmaxf(s_red[tid], s_red[tid + off]);
        __syncthreads();
    }
    float mx = s_red[0]; __syncthreads();
    float e = (tid < 196) ? __expf(s_full[tid] - mx) : 0.f;
    s_red[tid] = e; __syncthreads();
    for (int off = 128; off > 0; off >>= 1) {
        if (tid < off) s_red[tid] += s_red[tid + off];
        __syncthreads();
    }
    float inv = 1.f / s_red[0]; __syncthreads();
    if (tid < 196) s_alpha[tid] = e * inv;
    __syncthreads();

    bool active = (lengths[b] - 1) > t;
    if (tid < 196)
        alphas_out[((size_t)b * 31 + t) * 196 + tid] = active ? s_alpha[tid] : 0.f;

    // z[f] = sum_l alpha[l] * feat_r[b,l,f]  (feat_r = flat reinterpretation, stride 512 in l)
    for (int f = tid; f < 512; f += 256) {
        const float* fp = features + (size_t)b * 196 * 512 + f;
        float acc = 0.f;
        for (int l = 0; l < 196; ++l) acc += s_alpha[l] * fp[l * 512];
        float z = acc * df[b * 1024 + 512 + f];      // fb (sigmoid already applied)
        xh[(size_t)b * 1536 + 512 + f] = f2bf(z);
    }
    // embedding gather into xh[:, 0:512]
    int cap = captions[b * 32 + t];
    for (int e2 = tid; e2 < 512; e2 += 256)
        xh[(size_t)b * 1536 + e2] = Wemb_bf[(size_t)cap * 512 + e2];
}

// ---------------------------------------------------------------------------
// LSTM elementwise: gates[64,2048] -> c_new, h_new; masked carry; bf16 copies.
// Masked h (bf16) written in place into xh[:,1024:1536] (stride 1536).
// ---------------------------------------------------------------------------
__global__ void lstm_elem_kernel(const float* __restrict__ gates, float* __restrict__ h,
                                 float* __restrict__ c, uint16_t* __restrict__ xh,
                                 uint16_t* __restrict__ hnew_bf, const int* __restrict__ lengths,
                                 int t) {
    int b = blockIdx.x, n = threadIdx.x;             // 64 x 512
    const float* g = gates + (size_t)b * 2048;
    float ig = g[n], fg = g[512 + n], gg = g[1024 + n], og = g[1536 + n];
    float cn = sigf(fg) * c[b * 512 + n] + sigf(ig) * tanhf(gg);
    float hn = sigf(og) * tanhf(cn);
    hnew_bf[b * 512 + n] = f2bf(hn);                 // vocab projection uses unmasked h_new
    bool active = (lengths[b] - 1) > t;
    float hv = active ? hn : h[b * 512 + n];
    float cv = active ? cn : c[b * 512 + n];
    h[b * 512 + n] = hv;
    c[b * 512 + n] = cv;
    xh[(size_t)b * 1536 + 1024 + n] = f2bf(hv);      // recurrent path uses masked h
}

// ---------------------------------------------------------------------------
// Host orchestration
// ---------------------------------------------------------------------------
extern "C" void kernel_launch(void* const* d_in, const int* in_sizes, int n_in,
                              void* d_out, int out_size, void* d_ws, size_t ws_size,
                              hipStream_t stream) {
    constexpr int B = 64, T = 32, F = 512, L = 196, E = 512, H = 512, A = 512, V = 10000;
    constexpr int Tm = T - 1;   // 31

    const float* features = (const float*)d_in[0];
    const int*   captions = (const int*)d_in[1];
    const int*   lengths  = (const int*)d_in[2];
    const float* W_emb    = (const float*)d_in[3];
    const float* W_enc    = (const float*)d_in[4];
    const float* b_enc    = (const float*)d_in[5];
    const float* W_dec    = (const float*)d_in[6];
    const float* b_dec    = (const float*)d_in[7];
    const float* W_full   = (const float*)d_in[8];
    const float* b_full   = (const float*)d_in[9];
    const float* W_fbeta  = (const float*)d_in[10];
    const float* b_fbeta  = (const float*)d_in[11];
    const float* W_ih     = (const float*)d_in[12];
    const float* W_hh     = (const float*)d_in[13];
    const float* b_ih     = (const float*)d_in[14];
    const float* b_hh     = (const float*)d_in[15];
    const float* W_out    = (const float*)d_in[16];
    const float* b_out    = (const float*)d_in[17];
    const float* W_init_h = (const float*)d_in[18];
    const float* b_init_h = (const float*)d_in[19];
    const float* W_init_c = (const float*)d_in[20];
    const float* b_init_c = (const float*)d_in[21];

    // workspace carve-up (256B aligned)
    char* ws = (char*)d_ws;
    size_t off = 0;
    auto carve = [&](size_t bytes) -> char* {
        char* p = ws + off;
        off += (bytes + 255) & ~(size_t)255;
        return p;
    };
    uint16_t* feat_bf   = (uint16_t*)carve((size_t)B * L * F * 2);
    uint16_t* Wemb_bf   = (uint16_t*)carve((size_t)V * E * 2);
    uint16_t* Wenc_bf   = (uint16_t*)carve((size_t)A * F * 2);
    uint16_t* Wdf_bf    = (uint16_t*)carve((size_t)(A + F) * H * 2);      // [W_dec ; W_fbeta]
    uint16_t* Wcat_bf   = (uint16_t*)carve((size_t)4 * H * (E + F + H) * 2); // [W_ih | W_hh]
    uint16_t* Wout_bf   = (uint16_t*)carve((size_t)V * H * 2);
    uint16_t* Winith_bf = (uint16_t*)carve((size_t)H * F * 2);
    uint16_t* Winitc_bf = (uint16_t*)carve((size_t)H * F * 2);
    uint16_t* mean_bf   = (uint16_t*)carve((size_t)B * F * 2);
    float*    b_df      = (float*)carve((size_t)(A + F) * 4);             // [b_dec ; b_fbeta]
    float*    en_att    = (float*)carve((size_t)B * L * A * 4);
    float*    h_st      = (float*)carve((size_t)B * H * 4);
    float*    c_st      = (float*)carve((size_t)B * H * 4);
    float*    df_buf    = (float*)carve((size_t)B * (A + F) * 4);         // [de | fb]
    float*    gates     = (float*)carve((size_t)B * 4 * H * 4);
    uint16_t* xh        = (uint16_t*)carve((size_t)B * (E + F + H) * 2);  // [emb | z | h]
    uint16_t* hnew_bf   = (uint16_t*)carve((size_t)B * H * 2);

    float* pred  = (float*)d_out;                          // [B, Tm, V]
    float* alph  = pred + (size_t)B * Tm * V;              // [B, Tm, L]

    auto cvt = [&](const float* s, uint16_t* d, size_t n) {
        cvt_bf16_kernel<<<(unsigned)((n + 255) / 256), 256, 0, stream>>>(s, d, (int)n);
    };
    auto gemm = [&](const uint16_t* Am, int lda, const uint16_t* Wm,
                    const float* bias, const float* bias2, float* Cm,
                    uint16_t* Cbf, int M, int N, int K, int ldc, int ldcb, int act,
                    const int* len, int step) {
        int tiles = (M / 16) * (N / 16);
        int blocks = (tiles * 32 + 255) / 256;
        wmma_gemm_kernel<<<blocks, 256, 0, stream>>>(Am, lda, Wm, bias, bias2, Cm, Cbf,
                                                     M, N, K, ldc, ldcb, act, len, step);
    };

    // ---- one-time conversions ----
    cvt(features, feat_bf,   (size_t)B * L * F);
    cvt(W_emb,    Wemb_bf,   (size_t)V * E);
    cvt(W_enc,    Wenc_bf,   (size_t)A * F);
    cvt(W_dec,    Wdf_bf,                    (size_t)A * H);   // rows 0..511
    cvt(W_fbeta,  Wdf_bf + (size_t)A * H,    (size_t)F * H);   // rows 512..1023
    cvt(W_out,    Wout_bf,   (size_t)V * H);
    cvt(W_init_h, Winith_bf, (size_t)H * F);
    cvt(W_init_c, Winitc_bf, (size_t)H * F);
    {   // W_cat = [W_ih | W_hh] concat along K -> [2048, 1536]
        int total = 4 * H * (E + F + H);
        cvt_concat_kernel<<<(total + 255) / 256, 256, 0, stream>>>(
            W_ih, E + F, W_hh, H, Wcat_bf, total);
    }
    concat_bias_kernel<<<4, 256, 0, stream>>>(b_dec, A, b_fbeta, F, b_df);

    // ---- init: mean, h0 (bf16 into xh[:,1024:]), c0, en_att ----
    mean_kernel<<<B, 512, 0, stream>>>(features, mean_bf);
    gemm(mean_bf, F, Winith_bf, b_init_h, nullptr, h_st, xh + 1024,
         B, H, F, H, E + F + H, 0, nullptr, 0);
    gemm(mean_bf, F, Winitc_bf, b_init_c, nullptr, c_st, nullptr,
         B, H, F, H, 0, 0, nullptr, 0);
    gemm(feat_bf, F, Wenc_bf, b_enc, nullptr, en_att, nullptr,
         B * L, A, F, A, 0, 0, nullptr, 0);

    // ---- recurrent steps (5 launches per step) ----
    const uint16_t* h_bf = xh + 1024;   // masked-h slice, row stride 1536
    for (int t = 0; t < Tm; ++t) {
        // [de | fb] = h * [W_dec ; W_fbeta]^T, sigmoid on cols >= 512
        gemm(h_bf, E + F + H, Wdf_bf, b_df, nullptr, df_buf, nullptr,
             B, A + F, H, A + F, 0, /*act=*/2, nullptr, 0);
        attention_kernel<<<B, 256, 0, stream>>>(en_att, df_buf, W_full, b_full, features,
                                                captions, lengths, Wemb_bf, xh, alph, t);
        // gates = [emb|z|h] * [W_ih|W_hh]^T + b_ih + b_hh   (single fused GEMM)
        gemm(xh, E + F + H, Wcat_bf, b_ih, b_hh, gates, nullptr,
             B, 4 * H, E + F + H, 4 * H, 0, 0, nullptr, 0);
        lstm_elem_kernel<<<B, 512, 0, stream>>>(gates, h_st, c_st, xh, hnew_bf, lengths, t);
        // predictions[:, t, :] = masked(h_new * W_out^T + b_out)
        gemm(hnew_bf, H, Wout_bf, b_out, nullptr, pred + (size_t)t * V, nullptr,
             B, V, H, Tm * V, 0, 0, lengths, t);
    }
}
